// CPUAttentionBaseline_89970974917189
// MI455X (gfx1250) — matrix operand order repaired and checked by reading, not compiled, not run on hardware
//
#include <hip/hip_runtime.h>

#define B_  2
#define S_  2048
#define H_  1024
#define NH_ 16
#define HD_ 64

typedef __bf16 bf16;
typedef __attribute__((ext_vector_type(16))) __bf16 v16bf;
typedef __attribute__((ext_vector_type(8)))  float  v8f;

__device__ __forceinline__ v8f vzero8() {
  v8f z;
#pragma unroll
  for (int i = 0; i < 8; ++i) z[i] = 0.f;
  return z;
}

// Low 32 bits of a flat shared-aperture address are the LDS byte offset.
__device__ __forceinline__ unsigned lds_off(const void* p) {
  return (unsigned)(uintptr_t)p;
}
// GLOBAL_LOAD_ASYNC_TO_LDS_B128, GV mode (64-bit vaddr, saddr=off). ASYNCcnt.
__device__ __forceinline__ void async_b128(unsigned lds, const void* g) {
  asm volatile("global_load_async_to_lds_b128 %0, %1, off"
               :: "v"(lds), "v"((unsigned long long)(uintptr_t)g)
               : "memory");
}
__device__ __forceinline__ void wait_async0() {
  asm volatile("s_wait_asynccnt 0" ::: "memory");
}

// ---- DPP16 all-reduce over each 16-lane half: one VALU op per butterfly step.
// v = op(dpp(v), v)  with DPP on SRC0.
__device__ __forceinline__ float dpp_bcast_max16(float x) {
  asm("v_max_num_f32_dpp %0, %0, %0 quad_perm:[1,0,3,2] row_mask:0xf bank_mask:0xf bound_ctrl:1" : "+v"(x));
  asm("v_max_num_f32_dpp %0, %0, %0 quad_perm:[2,3,0,1] row_mask:0xf bank_mask:0xf bound_ctrl:1" : "+v"(x));
  asm("v_max_num_f32_dpp %0, %0, %0 row_ror:4 row_mask:0xf bank_mask:0xf bound_ctrl:1" : "+v"(x));
  asm("v_max_num_f32_dpp %0, %0, %0 row_ror:8 row_mask:0xf bank_mask:0xf bound_ctrl:1" : "+v"(x));
  return x;
}
__device__ __forceinline__ float dpp_bcast_add16(float x) {
  asm("v_add_f32_dpp %0, %0, %0 quad_perm:[1,0,3,2] row_mask:0xf bank_mask:0xf bound_ctrl:1" : "+v"(x));
  asm("v_add_f32_dpp %0, %0, %0 quad_perm:[2,3,0,1] row_mask:0xf bank_mask:0xf bound_ctrl:1" : "+v"(x));
  asm("v_add_f32_dpp %0, %0, %0 row_ror:4 row_mask:0xf bank_mask:0xf bound_ctrl:1" : "+v"(x));
  asm("v_add_f32_dpp %0, %0, %0 row_ror:8 row_mask:0xf bank_mask:0xf bound_ctrl:1" : "+v"(x));
  return x;
}

// ---------------------------------------------------------------- convert
__global__ void cvt_f32_bf16(const float* __restrict__ src, bf16* __restrict__ dst, int n) {
  int i = blockIdx.x * blockDim.x + threadIdx.x;
  int stride = gridDim.x * blockDim.x;
  for (; i < n; i += stride) dst[i] = (bf16)src[i];
}

// ---------------------------------------------------------------- GEMM  C = A * W^T
// A: [M,K] bf16 row-major, W: [N,K] bf16 row-major.
// MODE 0: scatter into Q[b,h,s,d], K[b,h,s,d], Vt[b,h,d,s] (bf16)
// MODE 1: write f32 C[M,N] row-major
template <int MODE>
__global__ __launch_bounds__(256) void gemm_bf16_wmma(
    const bf16* __restrict__ A, const bf16* __restrict__ W,
    void* __restrict__ out0, void* __restrict__ out1, void* __restrict__ out2,
    int M, int N, int K) {
  constexpr int LDP = 48;  // padded row in bf16 elems (96B, multiple of 16B)
  __shared__ __attribute__((aligned(16))) bf16 Asub[2][128][LDP];
  __shared__ __attribute__((aligned(16))) bf16 Bsub[2][128][LDP];

  const int tid   = threadIdx.x;
  const int wave  = tid >> 5;
  const int lane  = tid & 31;
  const int wm    = wave & 3;   // 0..3 -> 32 rows each
  const int wn    = wave >> 2;  // 0..1 -> 64 cols each
  const int m0    = blockIdx.y * 128;
  const int n0    = blockIdx.x * 128;
  const int lhalf = lane >> 4;
  const int l16   = lane & 15;

  v8f acc[2][4];
#pragma unroll
  for (int i = 0; i < 2; ++i)
#pragma unroll
    for (int j = 0; j < 4; ++j) acc[i][j] = vzero8();

  const int NK = K / 32;

  // per-thread load slots: 128x32 bf16 tile = 512 16B-chunks, 2 per thread
  const int c0 = tid * 2, c1 = tid * 2 + 1;
  const int r0 = c0 >> 2, h0 = c0 & 3;
  const int r1 = c1 >> 2, h1 = c1 & 3;

  // DMA one k-tile (A and B) straight into LDS buffer `buf`
  auto gasync = [&](int kk, int buf) {
    async_b128(lds_off(&Asub[buf][r0][h0 * 8]), A + (size_t)(m0 + r0) * K + kk * 32 + h0 * 8);
    async_b128(lds_off(&Asub[buf][r1][h1 * 8]), A + (size_t)(m0 + r1) * K + kk * 32 + h1 * 8);
    async_b128(lds_off(&Bsub[buf][r0][h0 * 8]), W + (size_t)(n0 + r0) * K + kk * 32 + h0 * 8);
    async_b128(lds_off(&Bsub[buf][r1][h1 * 8]), W + (size_t)(n0 + r1) * K + kk * 32 + h1 * 8);
  };

  gasync(0, 0);
  wait_async0();
  __syncthreads();

  for (int kk = 0; kk < NK; ++kk) {
    const int cur = kk & 1;
    if (kk + 1 < NK) gasync(kk + 1, cur ^ 1);  // overlaps with compute below

    v16bf afrag[2], bfrag[4];
#pragma unroll
    for (int tm = 0; tm < 2; ++tm) {
      const bf16* ap = &Asub[cur][wm * 32 + tm * 16 + l16][lhalf * 8];
      ((uint4*)&afrag[tm])[0] = *(const uint4*)ap;        // K = kb..kb+7
      ((uint4*)&afrag[tm])[1] = *(const uint4*)(ap + 16); // K = kb+16..kb+23
    }
#pragma unroll
    for (int tn = 0; tn < 4; ++tn) {
      const bf16* bp = &Bsub[cur][wn * 64 + tn * 16 + l16][lhalf * 8];
      ((uint4*)&bfrag[tn])[0] = *(const uint4*)bp;        // K = kb..kb+7
      ((uint4*)&bfrag[tn])[1] = *(const uint4*)(bp + 16); // K = kb+16..kb+23
    }
#pragma unroll
    for (int tm = 0; tm < 2; ++tm)
#pragma unroll
      for (int tn = 0; tn < 4; ++tn)
        acc[tm][tn] = __builtin_amdgcn_wmma_f32_16x16x32_bf16(
            false, afrag[tm], false, bfrag[tn], (short)0, acc[tm][tn], false, false);

    wait_async0();
    __syncthreads();
  }

#pragma unroll
  for (int tm = 0; tm < 2; ++tm)
#pragma unroll
    for (int tn = 0; tn < 4; ++tn)
#pragma unroll
      for (int i = 0; i < 8; ++i) {
        const int r = m0 + wm * 32 + tm * 16 + lhalf * 8 + i;
        const int o = n0 + wn * 64 + tn * 16 + l16;
        const float v = acc[tm][tn][i];
        if (MODE == 0) {
          const int bb = r >> 11, s = r & (S_ - 1);
          const int type = o >> 10, hh = (o >> 6) & 15, d = o & 63;
          const size_t bh = (size_t)(bb * NH_ + hh);
          if (type == 0)
            ((bf16*)out0)[(bh * S_ + s) * HD_ + d] = (bf16)v;
          else if (type == 1)
            ((bf16*)out1)[(bh * S_ + s) * HD_ + d] = (bf16)v;
          else
            ((bf16*)out2)[(bh * HD_ + d) * S_ + s] = (bf16)v;
        } else {
          ((float*)out0)[(size_t)r * N + o] = v;
        }
      }
}

// ---------------------------------------------------------------- flash attention
// 4 waves / WG; wave owns 16 query rows; 64-wide KV tiles double buffered via
// async DMA to LDS; DPP16 softmax reductions; exp in log2 domain.
__global__ __launch_bounds__(128) void attn_fa(
    const bf16* __restrict__ Q, const bf16* __restrict__ Km,
    const bf16* __restrict__ Vt, bf16* __restrict__ Obf) {
  __shared__ __attribute__((aligned(16))) bf16 Ksub[2][64][72];  // [t][d]
  __shared__ __attribute__((aligned(16))) bf16 Vsub[2][64][72];  // [d][t]
  __shared__ __attribute__((aligned(16))) bf16 Psub[4][16][72];  // per-wave P stage

  const int tid = threadIdx.x, wave = tid >> 5, lane = tid & 31;
  const int l16 = lane & 15, lhalf = lane >> 4;
  const int bh = blockIdx.z * NH_ + blockIdx.y;
  const int q0 = blockIdx.x * 64 + wave * 16;

  const bf16* Qbase = Q + (size_t)bh * S_ * HD_;
  const bf16* Kbase = Km + (size_t)bh * S_ * HD_;
  const bf16* Vbase = Vt + (size_t)bh * HD_ * S_;

  // Q fragments, pre-scaled by (1/sqrt(64)) * log2(e): scores land in log2 domain
  v16bf qf[2];
  {
    const bf16* qp = Qbase + (size_t)(q0 + l16) * HD_ + lhalf * 8;
    uint4 t[4];
    t[0] = *(const uint4*)(qp);
    t[1] = *(const uint4*)(qp + 16);
    t[2] = *(const uint4*)(qp + 32);
    t[3] = *(const uint4*)(qp + 48);
    bf16* e = (bf16*)t;
    const float sc = 0.125f * 1.44269504088896f;
#pragma unroll
    for (int i = 0; i < 32; ++i) e[i] = (bf16)((float)e[i] * sc);
    ((uint4*)&qf[0])[0] = t[0];
    ((uint4*)&qf[0])[1] = t[1];
    ((uint4*)&qf[1])[0] = t[2];
    ((uint4*)&qf[1])[1] = t[3];
  }

  v8f oacc[4];
#pragma unroll
  for (int tn = 0; tn < 4; ++tn) oacc[tn] = vzero8();
  float mrun[8], lrun[8];
#pragma unroll
  for (int i = 0; i < 8; ++i) { mrun[i] = -1e30f; lrun[i] = 0.f; }

  // DMA one 64-wide K tile + V^T tile into LDS buffer `buf`
  auto loadKV = [&](int kt, int buf) {
    const int t0 = kt * 64;
#pragma unroll
    for (int j = 0; j < 4; ++j) {
      const int c = tid + j * 128;          // 0..511 chunk id
      const int row = c >> 3, h = c & 7;
      async_b128(lds_off(&Ksub[buf][row][h * 8]),
                 Kbase + (size_t)(t0 + row) * HD_ + h * 8);
      async_b128(lds_off(&Vsub[buf][row][h * 8]),
                 Vbase + (size_t)row * S_ + t0 + h * 8);
    }
  };

  loadKV(0, 0);
  wait_async0();
  __syncthreads();

  const int NT = S_ / 64;
  for (int kt = 0; kt < NT; ++kt) {
    const int cur = kt & 1;
    if (kt + 1 < NT) loadKV(kt + 1, cur ^ 1);  // DMA overlaps compute

    // ---- scores S = Q*K^T (log2-scaled), 16x64 per wave
    v8f sacc[4];
#pragma unroll
    for (int tn = 0; tn < 4; ++tn) {
      sacc[tn] = vzero8();
      const bf16* kp = &Ksub[cur][tn * 16 + l16][lhalf * 8];
      v16bf b0, b1;
      ((uint4*)&b0)[0] = *(const uint4*)kp;
      ((uint4*)&b0)[1] = *(const uint4*)(kp + 16);
      ((uint4*)&b1)[0] = *(const uint4*)(kp + 32);
      ((uint4*)&b1)[1] = *(const uint4*)(kp + 48);
      sacc[tn] = __builtin_amdgcn_wmma_f32_16x16x32_bf16(false, qf[0], false, b0,
                                                         (short)0, sacc[tn], false, false);
      sacc[tn] = __builtin_amdgcn_wmma_f32_16x16x32_bf16(false, qf[1], false, b1,
                                                         (short)0, sacc[tn], false, false);
    }

    // ---- online softmax in log2 domain; row spans the 16 lanes of each half
#pragma unroll
    for (int i = 0; i < 8; ++i) {
      float mx = fmaxf(fmaxf(sacc[0][i], sacc[1][i]), fmaxf(sacc[2][i], sacc[3][i]));
      mx = dpp_bcast_max16(mx);
      const float mnew = fmaxf(mrun[i], mx);
      float s = 0.f;
#pragma unroll
      for (int tn = 0; tn < 4; ++tn) {
        const float p = __builtin_amdgcn_exp2f(sacc[tn][i] - mnew);
        sacc[tn][i] = p;
        s += p;
      }
      s = dpp_bcast_add16(s);
      const float alpha = __builtin_amdgcn_exp2f(mrun[i] - mnew);
      lrun[i] = lrun[i] * alpha + s;
      mrun[i] = mnew;
#pragma unroll
      for (int tn = 0; tn < 4; ++tn) oacc[tn][i] *= alpha;
    }

    // ---- P: accumulator layout -> LDS -> A-fragment layout (per-wave region)
#pragma unroll
    for (int tn = 0; tn < 4; ++tn)
#pragma unroll
      for (int i = 0; i < 8; ++i)
        Psub[wave][lhalf * 8 + i][tn * 16 + l16] = (bf16)sacc[tn][i];

    v16bf pf[2];
    {
      const bf16* pp = &Psub[wave][l16][lhalf * 8];
      ((uint4*)&pf[0])[0] = *(const uint4*)pp;
      ((uint4*)&pf[0])[1] = *(const uint4*)(pp + 16);
      ((uint4*)&pf[1])[0] = *(const uint4*)(pp + 32);
      ((uint4*)&pf[1])[1] = *(const uint4*)(pp + 48);
    }

    // ---- O += P * V   (V^T tile is K-contiguous)
#pragma unroll
    for (int tn = 0; tn < 4; ++tn) {
      const bf16* vp = &Vsub[cur][tn * 16 + l16][lhalf * 8];
      v16bf b0, b1;
      ((uint4*)&b0)[0] = *(const uint4*)vp;
      ((uint4*)&b0)[1] = *(const uint4*)(vp + 16);
      ((uint4*)&b1)[0] = *(const uint4*)(vp + 32);
      ((uint4*)&b1)[1] = *(const uint4*)(vp + 48);
      oacc[tn] = __builtin_amdgcn_wmma_f32_16x16x32_bf16(false, pf[0], false, b0,
                                                         (short)0, oacc[tn], false, false);
      oacc[tn] = __builtin_amdgcn_wmma_f32_16x16x32_bf16(false, pf[1], false, b1,
                                                         (short)0, oacc[tn], false, false);
    }

    wait_async0();
    __syncthreads();
  }

  // ---- normalize and emit attn output as bf16 [b][s][h*64+d]
#pragma unroll
  for (int tn = 0; tn < 4; ++tn)
#pragma unroll
    for (int i = 0; i < 8; ++i) {
      const int s = q0 + lhalf * 8 + i;
      const int d = tn * 16 + l16;
      const float v = oacc[tn][i] / lrun[i];
      Obf[((size_t)blockIdx.z * S_ + s) * H_ + blockIdx.y * HD_ + d] = (bf16)v;
    }
}

// ---------------------------------------------------------------- launcher
extern "C" void kernel_launch(void* const* d_in, const int* in_sizes, int n_in,
                              void* d_out, int out_size, void* d_ws, size_t ws_size,
                              hipStream_t stream) {
  const float* x    = (const float*)d_in[0];
  const float* wqkv = (const float*)d_in[1];
  const float* wout = (const float*)d_in[2];

  char* ws = (char*)d_ws;
  const size_t MB = 1u << 20;
  bf16* x_bf    = (bf16*)(ws + 0 * MB);    // 8 MB : [4096,1024]
  bf16* wqkv_bf = (bf16*)(ws + 8 * MB);    // 6 MB : [3072,1024]
  bf16* wout_bf = (bf16*)(ws + 14 * MB);   // 2 MB : [1024,1024]
  bf16* Qw      = (bf16*)(ws + 16 * MB);   // 8 MB : [b,h,s,d]
  bf16* Kw      = (bf16*)(ws + 24 * MB);   // 8 MB : [b,h,s,d]
  bf16* Vtw     = (bf16*)(ws + 32 * MB);   // 8 MB : [b,h,d,s]
  bf16* Abf     = (bf16*)(ws + 40 * MB);   // 8 MB : [4096,1024] attn out

  cvt_f32_bf16<<<2048, 256, 0, stream>>>(x, x_bf, B_ * S_ * H_);
  cvt_f32_bf16<<<2048, 256, 0, stream>>>(wqkv, wqkv_bf, 3 * H_ * H_);
  cvt_f32_bf16<<<1024, 256, 0, stream>>>(wout, wout_bf, H_ * H_);

  dim3 g0(3 * H_ / 128, B_ * S_ / 128);  // 24 x 32
  gemm_bf16_wmma<0><<<g0, 256, 0, stream>>>(x_bf, wqkv_bf, Qw, Kw, Vtw,
                                            B_ * S_, 3 * H_, H_);

  dim3 ga(S_ / 64, NH_, B_);             // 32 x 16 x 2
  attn_fa<<<ga, 128, 0, stream>>>(Qw, Kw, Vtw, Abf);

  dim3 g1(H_ / 128, B_ * S_ / 128);      // 8 x 32
  gemm_bf16_wmma<1><<<g1, 256, 0, stream>>>(Abf, wout_bf, d_out, nullptr, nullptr,
                                            B_ * S_, H_, H_);
}
